// HyperbolicContrastiveLoss_40518721470621
// MI455X (gfx1250) — matrix operand order repaired and checked by reading, not compiled, not run on hardware
//
#include <hip/hip_runtime.h>

// ---------------- constants from the reference ----------------
#define TEMP_INV     10.0f                  // 1 / TEMPERATURE
#define CURV         0.1f
#define SQRT_C       0.31622776601683794f   // sqrt(0.1)
#define INV_SQRT_C   3.1622776601683795f
#define SEP_W        0.1f
#define BND_W        0.01f
#define MIN_PD       2.0f
#define TGT_NORM     0.9f
#define EPS_ART      1e-5f
#define NEG_HUGE    -3.0e38f
#define DIM          512
#define AROW         1048   // ushorts per LDS row: [520 hi | 520 lo | 8 pad]
                            // 524 dwords == 12 mod 64 -> 16 distinct banks for m=0..15
#define ALO          520    // lo-half offset within a row (1040 B, 16B aligned)
#define ROWS         64     // rows per block in main kernel

typedef __attribute__((ext_vector_type(8)))  float  v8f_t;
typedef __attribute__((ext_vector_type(8)))  __bf16 v8bf_t;
typedef __attribute__((ext_vector_type(16))) __bf16 v16bf_t;

__device__ __forceinline__ v8f_t wmma_bf16(v16bf_t a, v16bf_t b, v8f_t c) {
    // D = A(16x32,bf16) * B(32x16,bf16) + C(16x16,f32), wave32
    return __builtin_amdgcn_wmma_f32_16x16x32_bf16(false, a, false, b,
                                                   (short)0, c, false, false);
}

__device__ __forceinline__ v16bf_t cat8(v8bf_t a, v8bf_t b) {
    return __builtin_shufflevector(a, b, 0,1,2,3,4,5,6,7,8,9,10,11,12,13,14,15);
}

// A fragment for one 16-row tile: K in [ka,ka+16) (contiguous, matches B)
__device__ __forceinline__ v16bf_t afrag(const unsigned short* base, int off) {
    return cat8(*(const v8bf_t*)(base + off), *(const v8bf_t*)(base + off + 8));
}

// split f32 into bf16 hi (truncated) + bf16 lo of the residual; x ~= hi + lo + O(2^-16 x)
__device__ __forceinline__ void bf16_split(float x, unsigned short& hi, unsigned short& lo) {
    unsigned int b = __float_as_uint(x);
    hi = (unsigned short)(b >> 16);
    float r = x - __uint_as_float(b & 0xFFFF0000u);
    lo = (unsigned short)(__float_as_uint(r) >> 16);
}

// Poincare distance from uv = <-x,y>, x2 = |x|^2, y2 = |y|^2
__device__ __forceinline__ float poincare_dist(float uv, float x2, float y2) {
    float a      = 1.0f + 2.0f * CURV * uv + CURV * y2;
    float b      = 1.0f - CURV * x2;
    float denom  = 1.0f + 2.0f * CURV * uv + CURV * CURV * x2 * y2;
    float num_sq = fmaxf(a * a * x2 + 2.0f * a * b * uv + b * b * y2, 0.0f);
    float arg    = SQRT_C * sqrtf(num_sq) / fmaxf(denom, 1e-15f);
    float z      = fminf(fmaxf(arg, -1.0f + EPS_ART), 1.0f - EPS_ART);
    return INV_SQRT_C * log1pf(2.0f * z / (1.0f - z));
}

// ---------------- squared row norms for embeddings + prototypes ----------------
__global__ __launch_bounds__(256) void sqnorm_kernel(
    const float* __restrict__ emb, const float* __restrict__ protos,
    int M, int K, float* __restrict__ x2, float* __restrict__ y2)
{
    const int wave = threadIdx.x >> 5, lane = threadIdx.x & 31;
    const int row  = blockIdx.x * 8 + wave;
    const float* src; float* dst; int idx;
    if (row < M)          { src = emb    + (size_t)row * DIM;       dst = x2; idx = row; }
    else if (row < M + K) { src = protos + (size_t)(row - M) * DIM; dst = y2; idx = row - M; }
    else return;
    float s = 0.0f;
    #pragma unroll
    for (int j = lane * 4; j < DIM; j += 128) {
        float4 v = *(const float4*)(src + j);
        s += v.x * v.x + v.y * v.y + v.z * v.z + v.w * v.w;
    }
    #pragma unroll
    for (int m = 16; m >= 1; m >>= 1) s += __shfl_xor(s, m, 32);
    if (lane == 0) dst[idx] = s;
}

// ---------------- prototypes f32 -> bf16 hi/lo (packed once, reused by both GEMMs) ----
__global__ __launch_bounds__(256) void cvt_protos_kernel(
    const float* __restrict__ protos, int total4,
    unsigned short* __restrict__ phi, unsigned short* __restrict__ plo)
{
    int i = blockIdx.x * 256 + threadIdx.x;   // float4 index
    if (i >= total4) return;
    float4 v = ((const float4*)protos)[i];
    ushort4 h, l;
    bf16_split(v.x, h.x, l.x);
    bf16_split(v.y, h.y, l.y);
    bf16_split(v.z, h.z, l.z);
    bf16_split(v.w, h.w, l.w);
    ((ushort4*)phi)[i] = h;
    ((ushort4*)plo)[i] = l;
}

// ---------------- main fused GEMM + distance + logsumexp kernel ----------------
// One block = 64 rows x all K cols. 8 waves; wave w owns col tiles w, w+8, ...
// and all 4 row tiles (12 WMMAs per B-fragment pair -> halves L2 traffic on protos).
// A hi/lo interleaved per LDS row so all fragment loads use 2 bases + DS immediates.
// Running softmax stats live in LDS (updated by lanes 0/16) to stay under 256 VGPRs.
// GEMM runs as bf16 split-precision: dot = hh + hl + lh (error ~2^-16, << f32 tolerance)
__global__ __launch_bounds__(256) void hyp_contrastive_kernel(
    const float* __restrict__ emb, const int* __restrict__ labels,
    const unsigned short* __restrict__ phi, const unsigned short* __restrict__ plo,
    const float* __restrict__ x2g, const float* __restrict__ y2g, int K,
    float* __restrict__ nllB, int* __restrict__ cntB)
{
    __shared__ unsigned short Abuf[ROWS * AROW];     // 131 KB (hi|lo interleaved per row)
    __shared__ float stats[8][ROWS][2];              // 4 KB  (running rowmax / rowsumexp)
    __shared__ float pickedL[ROWS];
    __shared__ int   labs[ROWS];
    __shared__ float rx2[ROWS];
    __shared__ float rnll[ROWS];
    __shared__ int   rcnt[ROWS];

    const int tid  = threadIdx.x;
    const int wave = tid >> 5, lane = tid & 31;
    const int half = lane >> 4, n = lane & 15;
    const int rowBase = blockIdx.x * ROWS;

    // stage A-strip: load 64x512 f32, split to bf16 hi/lo in LDS (32 float4 / thread)
    const float* xbase = emb + (size_t)rowBase * DIM;
    #pragma unroll 8
    for (int i = 0; i < 32; ++i) {
        int j = tid + i * 256;
        int r = j >> 7, c = (j & 127) << 2;
        float4 v = *(const float4*)(xbase + (size_t)r * DIM + c);
        ushort4 h, l;
        bf16_split(v.x, h.x, l.x);
        bf16_split(v.y, h.y, l.y);
        bf16_split(v.z, h.z, l.z);
        bf16_split(v.w, h.w, l.w);
        *(ushort4*)&Abuf[r * AROW + c]       = h;
        *(ushort4*)&Abuf[r * AROW + ALO + c] = l;
    }
    {   // init running stats in LDS
        float* sp = &stats[0][0][0];
        for (int i = tid; i < 8 * ROWS; i += 256) { sp[2*i] = NEG_HUGE; sp[2*i + 1] = 0.0f; }
    }
    if (tid < ROWS) {
        labs[tid]    = labels[rowBase + tid];
        rx2[tid]     = x2g[rowBase + tid];
        pickedL[tid] = 0.0f;
    }
    __syncthreads();

    // two LDS bases cover all 16 fragment loads per chunk via immediate offsets
    const unsigned short* A01 = &Abuf[ n       * AROW];   // rows n, n+16 (hi & lo)
    const unsigned short* A23 = &Abuf[(n + 32) * AROW];   // rows n+32, n+48 (hi & lo)
    const int R16 = 16 * AROW;

    const int numTiles = K >> 4;
    for (int ct = wave; ct < numTiles; ct += 8) {
        const int colBase = ct << 4;
        const int col = colBase + n;
        const unsigned short* Bh = phi + (size_t)col * DIM;
        const unsigned short* Bl = plo + (size_t)col * DIM;

        v8f_t zero = {0.f,0.f,0.f,0.f,0.f,0.f,0.f,0.f};
        v8f_t acc[4]; acc[0] = zero; acc[1] = zero; acc[2] = zero; acc[3] = zero;

        #pragma unroll 2
        for (int kk = 0; kk < DIM; kk += 32) {
            const int ka = kk + (half << 4);        // A frag: K in [ka,ka+16)
            const int kb = kk + (half << 4);        // B frag: K in [kb,kb+16)
            v16bf_t bh = *(const v16bf_t*)(Bh + kb);              // global 32B
            v16bf_t bl = *(const v16bf_t*)(Bl + kb);              // global 32B
            // row-tile pair 0/1 (limits live A-fragments to 4)
            {
                v16bf_t a0h = afrag(A01, ka);
                v16bf_t a1h = afrag(A01, R16 + ka);
                v16bf_t a0l = afrag(A01, ALO + ka);
                v16bf_t a1l = afrag(A01, R16 + ALO + ka);
                acc[0] = wmma_bf16(a0h, bh, acc[0]);
                acc[1] = wmma_bf16(a1h, bh, acc[1]);
                acc[0] = wmma_bf16(a0h, bl, acc[0]);
                acc[1] = wmma_bf16(a1h, bl, acc[1]);
                acc[0] = wmma_bf16(a0l, bh, acc[0]);
                acc[1] = wmma_bf16(a1l, bh, acc[1]);
            }
            // row-tile pair 2/3
            {
                v16bf_t a2h = afrag(A23, ka);
                v16bf_t a3h = afrag(A23, R16 + ka);
                v16bf_t a2l = afrag(A23, ALO + ka);
                v16bf_t a3l = afrag(A23, R16 + ALO + ka);
                acc[2] = wmma_bf16(a2h, bh, acc[2]);
                acc[3] = wmma_bf16(a3h, bh, acc[3]);
                acc[2] = wmma_bf16(a2h, bl, acc[2]);
                acc[3] = wmma_bf16(a3h, bl, acc[3]);
                acc[2] = wmma_bf16(a2l, bh, acc[2]);
                acc[3] = wmma_bf16(a3l, bh, acc[3]);
            }
        }

        const float ycol2 = y2g[col];
        #pragma unroll
        for (int t = 0; t < 4; ++t) {
            #pragma unroll
            for (int v = 0; v < 8; ++v) {
                const int lr  = v + (half << 3) + (t << 4);
                float dotv = acc[t][v];
                float lgt  = -poincare_dist(-dotv, rx2[lr], ycol2) * TEMP_INV;
                if (col == labs[lr]) pickedL[lr] = lgt;
                float tmax = lgt;
                tmax = fmaxf(tmax, __shfl_xor(tmax, 1, 32));
                tmax = fmaxf(tmax, __shfl_xor(tmax, 2, 32));
                tmax = fmaxf(tmax, __shfl_xor(tmax, 4, 32));
                tmax = fmaxf(tmax, __shfl_xor(tmax, 8, 32));
                float e = __expf(lgt - tmax);
                e += __shfl_xor(e, 1, 32);
                e += __shfl_xor(e, 2, 32);
                e += __shfl_xor(e, 4, 32);
                e += __shfl_xor(e, 8, 32);
                // online merge into LDS-resident per-wave stats (lanes 0 and 16 own
                // disjoint rows; no races)
                if ((lane & 15) == 0) {
                    float om = stats[wave][lr][0];
                    float os = stats[wave][lr][1];
                    float nm = fmaxf(om, tmax);
                    stats[wave][lr][0] = nm;
                    stats[wave][lr][1] = os * __expf(om - nm) + e * __expf(tmax - nm);
                }
            }
        }
    }
    __syncthreads();

    // merge 8 partial (max,sumexp) per row -> per-row nll
    if (tid < ROWS) {
        float Mx = NEG_HUGE;
        #pragma unroll
        for (int w = 0; w < 8; ++w) Mx = fmaxf(Mx, stats[w][tid][0]);
        float S = 0.0f;
        #pragma unroll
        for (int w = 0; w < 8; ++w) S += stats[w][tid][1] * __expf(stats[w][tid][0] - Mx);
        float logz = Mx + __logf(S);
        const int lb = labs[tid];
        const bool valid = lb >= 0;
        rnll[tid] = valid ? (logz - pickedL[tid]) : 0.0f;
        rcnt[tid] = valid ? 1 : 0;
    }
    __syncthreads();

    if (tid < 32) {
        float nll = rnll[tid] + rnll[tid + 32];
        int   cnt = rcnt[tid] + rcnt[tid + 32];
        #pragma unroll
        for (int m = 16; m >= 1; m >>= 1) {
            nll += __shfl_xor(nll, m, 32);
            cnt += __shfl_xor(cnt, m, 32);
        }
        if (tid == 0) { nllB[blockIdx.x] = nll; cntB[blockIdx.x] = cnt; }
    }
}

// ---------------- prototype separation: P x P^T, hinge on strict upper tri ----
__global__ __launch_bounds__(256) void proto_sep_kernel(
    const unsigned short* __restrict__ phi, const unsigned short* __restrict__ plo,
    const float* __restrict__ y2g, int K, float* __restrict__ sepB)
{
    __shared__ unsigned short Abuf[32 * AROW];
    __shared__ float wsum[8];
    const int tid  = threadIdx.x, wave = tid >> 5, lane = tid & 31;
    const int half = lane >> 4, n = lane & 15;
    const int rowBase = blockIdx.x * 32;

    // stage 32 prototype rows (already bf16) into LDS: 8 uint4 per thread per array
    #pragma unroll
    for (int i = 0; i < 8; ++i) {
        int j = tid + i * 256;
        int r = j >> 6, c = (j & 63) << 3;
        *(uint4*)&Abuf[r * AROW + c]       = *(const uint4*)&phi[(size_t)(rowBase + r) * DIM + c];
        *(uint4*)&Abuf[r * AROW + ALO + c] = *(const uint4*)&plo[(size_t)(rowBase + r) * DIM + c];
    }
    __syncthreads();

    const unsigned short* A01 = &Abuf[n * AROW];   // rows n, n+16 (hi & lo)
    const int R16 = 16 * AROW;

    float local = 0.0f;
    const int numTiles = K >> 4;
    for (int ct = wave; ct < numTiles; ct += 8) {
        const int colBase = ct << 4;
        if (colBase + 15 <= rowBase) continue;   // tile entirely at/below diagonal
        const int col = colBase + n;
        const unsigned short* Bh = phi + (size_t)col * DIM;
        const unsigned short* Bl = plo + (size_t)col * DIM;

        v8f_t zero = {0.f,0.f,0.f,0.f,0.f,0.f,0.f,0.f};
        v8f_t acc[2]; acc[0] = zero; acc[1] = zero;
        #pragma unroll 4
        for (int kk = 0; kk < DIM; kk += 32) {
            const int ka = kk + (half << 4);
            const int kb = kk + (half << 4);
            v16bf_t bh = *(const v16bf_t*)(Bh + kb);
            v16bf_t bl = *(const v16bf_t*)(Bl + kb);
            v16bf_t ah0 = afrag(A01, ka);
            v16bf_t ah1 = afrag(A01, R16 + ka);
            v16bf_t al0 = afrag(A01, ALO + ka);
            v16bf_t al1 = afrag(A01, R16 + ALO + ka);
            acc[0] = wmma_bf16(ah0, bh, acc[0]);
            acc[1] = wmma_bf16(ah1, bh, acc[1]);
            acc[0] = wmma_bf16(ah0, bl, acc[0]);
            acc[1] = wmma_bf16(ah1, bl, acc[1]);
            acc[0] = wmma_bf16(al0, bh, acc[0]);
            acc[1] = wmma_bf16(al1, bh, acc[1]);
        }
        const float ycol2 = y2g[col];
        #pragma unroll
        for (int t = 0; t < 2; ++t) {
            #pragma unroll
            for (int v = 0; v < 8; ++v) {
                const int lr = v + (half << 3) + (t << 4);
                const int gr = rowBase + lr;
                if (col > gr) {
                    float pd = poincare_dist(-acc[t][v], y2g[gr], ycol2);
                    local += fmaxf(MIN_PD - pd, 0.0f);
                }
            }
        }
    }
    #pragma unroll
    for (int m = 16; m >= 1; m >>= 1) local += __shfl_xor(local, m, 32);
    if (lane == 0) wsum[wave] = local;
    __syncthreads();
    if (tid == 0) {
        float s = 0.0f;
        for (int w = 0; w < 8; ++w) s += wsum[w];
        sepB[blockIdx.x] = s;
    }
}

// ---------------- boundary push on prototype norms ----------------
__global__ __launch_bounds__(256) void boundary_kernel(
    const float* __restrict__ y2g, int K, float* __restrict__ bndB)
{
    __shared__ float wsum[8];
    int i = blockIdx.x * 256 + threadIdx.x;
    float v = 0.0f;
    if (i < K) {
        float nrm = sqrtf(y2g[i]);
        v = fmaxf(TGT_NORM * INV_SQRT_C * 0.95f - nrm, 0.0f);
    }
    #pragma unroll
    for (int m = 16; m >= 1; m >>= 1) v += __shfl_xor(v, m, 32);
    if ((threadIdx.x & 31) == 0) wsum[threadIdx.x >> 5] = v;
    __syncthreads();
    if (threadIdx.x == 0) {
        float s = 0.0f;
        for (int w = 0; w < 8; ++w) s += wsum[w];
        bndB[blockIdx.x] = s;
    }
}

// ---------------- deterministic final reduction -> scalar loss ----------------
__global__ __launch_bounds__(256) void final_reduce_kernel(
    const float* __restrict__ nllB, const int* __restrict__ cntB, int nM,
    const float* __restrict__ sepB, int nS,
    const float* __restrict__ bndB, int nB,
    int K, float* __restrict__ out)
{
    __shared__ float sh[3][8];
    __shared__ int   shc[8];
    float nll = 0.0f, sep = 0.0f, bnd = 0.0f; int cnt = 0;
    for (int i = threadIdx.x; i < nM; i += 256) { nll += nllB[i]; cnt += cntB[i]; }
    for (int i = threadIdx.x; i < nS; i += 256) sep += sepB[i];
    for (int i = threadIdx.x; i < nB; i += 256) bnd += bndB[i];
    #pragma unroll
    for (int m = 16; m >= 1; m >>= 1) {
        nll += __shfl_xor(nll, m, 32);
        sep += __shfl_xor(sep, m, 32);
        bnd += __shfl_xor(bnd, m, 32);
        cnt += __shfl_xor(cnt, m, 32);
    }
    const int wave = threadIdx.x >> 5;
    if ((threadIdx.x & 31) == 0) { sh[0][wave] = nll; sh[1][wave] = sep; sh[2][wave] = bnd; shc[wave] = cnt; }
    __syncthreads();
    if (threadIdx.x == 0) {
        float N = 0.0f, S = 0.0f, Bd = 0.0f; int C = 0;
        for (int w = 0; w < 8; ++w) { N += sh[0][w]; S += sh[1][w]; Bd += sh[2][w]; C += shc[w]; }
        float npairs = 0.5f * (float)K * (float)(K - 1);
        out[0] = N / (float)(C < 1 ? 1 : C) + SEP_W * (S / npairs) + BND_W * (Bd / (float)K);
    }
}

extern "C" void kernel_launch(void* const* d_in, const int* in_sizes, int n_in,
                              void* d_out, int out_size, void* d_ws, size_t ws_size,
                              hipStream_t stream) {
    const float* emb    = (const float*)d_in[0];
    const int*   labels = (const int*)d_in[1];   // JAX demotes int64->int32 (x64 off)
    const float* protos = (const float*)d_in[2];
    const int M = in_sizes[0] / DIM;             // 32768
    const int K = in_sizes[2] / DIM;             // 1024

    // workspace: x2[M] | y2[K] | nllB | cntB | sepB | bndB | phi[K*DIM] | plo[K*DIM]
    float* x2   = (float*)d_ws;
    float* y2   = x2 + M;
    float* nllB = y2 + K;
    const int nMB = M / ROWS;
    int*   cntB = (int*)(nllB + nMB);
    const int nSB = K / 32;
    float* sepB = (float*)(cntB + nMB);
    const int nBB = (K + 255) / 256;
    float* bndB = sepB + nSB;
    unsigned long long p = (unsigned long long)(bndB + nBB);
    p = (p + 31ull) & ~31ull;
    unsigned short* phi = (unsigned short*)p;
    unsigned short* plo = phi + (size_t)K * DIM;

    const int total4 = K * DIM / 4;
    sqnorm_kernel<<<(M + K + 7) / 8, 256, 0, stream>>>(emb, protos, M, K, x2, y2);
    cvt_protos_kernel<<<(total4 + 255) / 256, 256, 0, stream>>>(protos, total4, phi, plo);
    hyp_contrastive_kernel<<<M / ROWS, 256, 0, stream>>>(emb, labels, phi, plo, x2, y2, K, nllB, cntB);
    proto_sep_kernel<<<K / 32, 256, 0, stream>>>(phi, plo, y2, K, sepB);
    boundary_kernel<<<nBB, 256, 0, stream>>>(y2, K, bndB);
    final_reduce_kernel<<<1, 256, 0, stream>>>(nllB, cntB, nMB, sepB, nSB, bndB, nBB, K, (float*)d_out);
}
